// self_attention_34849364639878
// MI455X (gfx1250) — matrix operand order repaired and checked by reading, not compiled, not run on hardware
//
#include <hip/hip_runtime.h>
#include <stdint.h>

typedef __attribute__((ext_vector_type(16))) __bf16 v16bf;
typedef __attribute__((ext_vector_type(8)))  float  v8f;

union BFV { uint32_t u[8]; v16bf v; };

__device__ __forceinline__ uint32_t f2bf_u(float f) {
    uint32_t u = __builtin_bit_cast(uint32_t, f);
    u += 0x7FFFu + ((u >> 16) & 1u);   // round-to-nearest-even
    return u >> 16;
}
__device__ __forceinline__ uint32_t pack2bf(float a, float b) {
    return f2bf_u(a) | (f2bf_u(b) << 16);
}
__device__ __forceinline__ uint32_t scale_pair(uint32_t w, float s) {
    float lo = __builtin_bit_cast(float, w << 16);
    float hi = __builtin_bit_cast(float, w & 0xffff0000u);
    return pack2bf(lo * s, hi * s);
}
__device__ __forceinline__ v16bf frag_from(uint4 a, uint4 b) {
    BFV t;
    t.u[0] = a.x; t.u[1] = a.y; t.u[2] = a.z; t.u[3] = a.w;
    t.u[4] = b.x; t.u[5] = b.y; t.u[6] = b.z; t.u[7] = b.w;
    return t.v;
}
__device__ __forceinline__ v8f wmma_bf16(v16bf a, v16bf b, v8f c) {
    return __builtin_amdgcn_wmma_f32_16x16x32_bf16(false, a, false, b, (short)0, c,
                                                   false, false);
}

// CDNA5 async DMA global->LDS (ASYNCcnt); GV addressing, 16B per lane.
#define ASYNC_B128(ldsoff, gptr)                                              \
    asm volatile("global_load_async_to_lds_b128 %0, %1, off"                  \
                 :: "v"(ldsoff), "v"((uint64_t)(uintptr_t)(gptr)) : "memory")
#define WAIT_ASYNC() asm volatile("s_wait_asynccnt 0x0" ::: "memory")

__device__ __forceinline__ uint32_t lds_off(const void* p) {
    return (uint32_t)(uintptr_t)p;   // low 32 bits of generic LDS ptr = DS offset
}

// ---------------------------------------------------------------------------
// Prep kernels: one-time fp32 -> bf16 convert / transpose-convert.
// ---------------------------------------------------------------------------
__global__ __launch_bounds__(256)
void conv_bf16_kernel(const float* __restrict__ in, uint16_t* __restrict__ out,
                      long n4)
{
    long i = (long)blockIdx.x * 256 + threadIdx.x;
    if (i < n4) {
        float4 f = *(const float4*)(in + i * 4);
        uint2 p; p.x = pack2bf(f.x, f.y); p.y = pack2bf(f.z, f.w);
        *(uint2*)(out + i * 4) = p;
    }
}

__global__ __launch_bounds__(256)
void transpose_conv_kernel(const float* __restrict__ in,  // [K][N]
                           uint16_t* __restrict__ out,    // [N][K] bf16
                           int K, int N)
{
    __shared__ float tile[32][33];
    const int bx = blockIdx.x * 32, by = blockIdx.y * 32;
    const int tx = threadIdx.x & 31, ty = threadIdx.x >> 5;   // 32 x 8
#pragma unroll
    for (int r = 0; r < 32; r += 8)
        tile[ty + r][tx] = in[(size_t)(by + ty + r) * N + bx + tx];
    __syncthreads();
#pragma unroll
    for (int r = 0; r < 32; r += 8)
        out[(size_t)(bx + ty + r) * K + by + tx] = (uint16_t)f2bf_u(tile[tx][ty + r]);
}

// ---------------------------------------------------------------------------
// GEMM: C = A[M,K](bf16) * Bt[N,K](bf16) + bias[N]
// MODE 0: fp32 out [M,N].
// MODE 1: qkv split -- cols <2048 to qk[M,2048] bf16 (token-major);
//         cols >=2048 to vT[1024][M] bf16 (dim-major, packed b128 stores).
// 128x128x32 tiles, 8 waves x (32x64), 2-step double-buffered async staging.
// ---------------------------------------------------------------------------
constexpr int BM = 128, BN = 128, BK = 32, LDA_ = 40, LDB_ = 40;

template <int MODE>
__global__ __launch_bounds__(256)
void gemm_bf16_kernel(const uint16_t* __restrict__ A,   // [M,K]
                      const uint16_t* __restrict__ Bt,  // [N,K]
                      const float* __restrict__ bias,
                      void* __restrict__ Cout, uint16_t* __restrict__ vTout,
                      int M, int N, int K)
{
    __shared__ __align__(16) uint16_t Al[2][BM * LDA_];
    __shared__ __align__(16) uint16_t Bl[2][BN * LDB_];

    const int tid  = threadIdx.x;
    const int wave = tid >> 5, lane = tid & 31;
    const int l15  = lane & 15, hlf = lane >> 4;
    const int m0 = blockIdx.y * BM, n0 = blockIdx.x * BN;
    const int wm = (wave & 3) * 32, wn = (wave >> 2) * 64;
    const int klo = hlf * 8, kb = hlf * 16;

    const v8f vzero = {0.f, 0.f, 0.f, 0.f, 0.f, 0.f, 0.f, 0.f};
    v8f acc[2][4];
    for (int i = 0; i < 2; ++i)
        for (int j = 0; j < 4; ++j) acc[i][j] = vzero;

    const int r0 = tid >> 2, c0 = (tid & 3) * 8;   // 16B chunk per lane

    auto stage = [&](int k0, int buf) {
        ASYNC_B128(lds_off(&Al[buf][r0 * LDA_ + c0]),
                   A + (size_t)(m0 + r0) * K + k0 + c0);
        ASYNC_B128(lds_off(&Al[buf][(r0 + 64) * LDA_ + c0]),
                   A + (size_t)(m0 + r0 + 64) * K + k0 + c0);
        ASYNC_B128(lds_off(&Bl[buf][r0 * LDB_ + c0]),
                   Bt + (size_t)(n0 + r0) * K + k0 + c0);
        ASYNC_B128(lds_off(&Bl[buf][(r0 + 64) * LDB_ + c0]),
                   Bt + (size_t)(n0 + r0 + 64) * K + k0 + c0);
    };
    auto compute_step = [&](int buf) {
        v16bf af[2], bfr[4];
#pragma unroll
        for (int fi = 0; fi < 2; ++fi) {
            const uint16_t* p = &Al[buf][(wm + fi * 16 + l15) * LDA_ + kb];
            af[fi] = frag_from(*(const uint4*)p, *(const uint4*)(p + 8));
        }
#pragma unroll
        for (int fj = 0; fj < 4; ++fj) {
            const uint16_t* p = &Bl[buf][(wn + fj * 16 + l15) * LDB_ + kb];
            bfr[fj] = frag_from(*(const uint4*)p, *(const uint4*)(p + 8));
        }
#pragma unroll
        for (int fi = 0; fi < 2; ++fi)
#pragma unroll
            for (int fj = 0; fj < 4; ++fj)
                acc[fi][fj] = wmma_bf16(af[fi], bfr[fj], acc[fi][fj]);
    };

    stage(0, 0);
    WAIT_ASYNC();
    __syncthreads();

    for (int k0 = 0; k0 < K; k0 += 2 * BK) {   // K % (2*BK) == 0
        stage(k0 + BK, 1);
        compute_step(0);
        WAIT_ASYNC();
        __syncthreads();
        if (k0 + 2 * BK < K) stage(k0 + 2 * BK, 0);
        compute_step(1);
        WAIT_ASYNC();
        __syncthreads();
    }

    // ---- epilogue (C layout: VGPR i -> rows i / i+8 by lane half)
    if (MODE == 1 && n0 >= 2048) {
        // V region: 8 consecutive tokens per lane are contiguous in vT -> b128
#pragma unroll
        for (int fi = 0; fi < 2; ++fi) {
#pragma unroll
            for (int fj = 0; fj < 4; ++fj) {
                int col = n0 + wn + fj * 16 + l15;
                float bv = bias[col];
                int rowbase = m0 + wm + fi * 16 + hlf * 8;   // token of element 0
                uint4 pk;
                pk.x = pack2bf(acc[fi][fj][0] + bv, acc[fi][fj][1] + bv);
                pk.y = pack2bf(acc[fi][fj][2] + bv, acc[fi][fj][3] + bv);
                pk.z = pack2bf(acc[fi][fj][4] + bv, acc[fi][fj][5] + bv);
                pk.w = pack2bf(acc[fi][fj][6] + bv, acc[fi][fj][7] + bv);
                *(uint4*)(vTout + (size_t)(col - 2048) * M + rowbase) = pk;
            }
        }
    } else {
#pragma unroll
        for (int fi = 0; fi < 2; ++fi) {
#pragma unroll
            for (int fj = 0; fj < 4; ++fj) {
                int col = n0 + wn + fj * 16 + l15;
                float bv = bias[col];
#pragma unroll
                for (int i = 0; i < 8; ++i) {
                    int row = m0 + wm + fi * 16 + i + hlf * 8;
                    float v = acc[fi][fj][i] + bv;
                    if (MODE == 1)   // Q/K region, token-major, stride 2048
                        ((uint16_t*)Cout)[(size_t)row * 2048 + col] =
                            (uint16_t)f2bf_u(v);
                    else
                        ((float*)Cout)[(size_t)row * N + col] = v;
                }
            }
        }
    }
}

// ---------------------------------------------------------------------------
// Flash attention: block = (b, h, 128-query slab); 8 waves x 16 query rows.
// 64-key chunks, all staging via async DMA, WMMA row sums.
// Keys are processed in a permuted order (free, via async LDS addressing):
// logical key k -> physical K row (k>>2)+(k&3)*16, so score tile t / col l15
// covers logical key 4*l15+t and a lane's four P values pack into one b64.
// The induced P-slot order is the identity, so V staging stays contiguous.
// qk bf16 [B*S, 2048]: Q cols 0..1023, K cols 1024..2047 (head h at h*64).
// vT bf16 [1024][B*S]: V dim-major. Output bf16.
// ---------------------------------------------------------------------------
__global__ __launch_bounds__(256)
void attn_kernel(const uint16_t* __restrict__ qk, const uint16_t* __restrict__ vT,
                 uint16_t* __restrict__ out)
{
    constexpr int S = 2048, D2 = 2048, D = 1024, NT = 8192;  // NT = total tokens
    constexpr int CK = 64;                        // keys per chunk
    constexpr int LDK = 72, LDV = 72, LDP = 72;
    __shared__ __align__(16) uint16_t Kl[CK * LDK];        // [perm key][dim]
    __shared__ __align__(16) uint16_t Vl[64 * LDV];        // [dim][key]
    __shared__ __align__(16) uint16_t Pl[8 * 16 * LDP];    // per-wave P scratch

    const int tid = threadIdx.x, wave = tid >> 5, lane = tid & 31;
    const int l15 = lane & 15, hlf = lane >> 4;
    const int qblk = blockIdx.x, h = blockIdx.y, b = blockIdx.z;
    const int q0 = qblk * 128 + wave * 16;
    const int klo = hlf * 8, kb = hlf * 16;
    uint16_t* Pw = Pl + wave * 16 * LDP;

    // Q fragments (A layout), pre-scaled by 1/sqrt(Dh) = 0.125
    v16bf qf[2];
    {
        const uint16_t* p = qk + ((size_t)(b * S + q0 + l15)) * D2 + h * 64;
#pragma unroll
        for (int dc = 0; dc < 2; ++dc) {
            const uint16_t* pp = p + dc * 32 + kb;
            uint4 a = *(const uint4*)pp, bq = *(const uint4*)(pp + 8);
            BFV t;
            t.u[0] = scale_pair(a.x, 0.125f);  t.u[1] = scale_pair(a.y, 0.125f);
            t.u[2] = scale_pair(a.z, 0.125f);  t.u[3] = scale_pair(a.w, 0.125f);
            t.u[4] = scale_pair(bq.x, 0.125f); t.u[5] = scale_pair(bq.y, 0.125f);
            t.u[6] = scale_pair(bq.z, 0.125f); t.u[7] = scale_pair(bq.w, 0.125f);
            qf[dc] = t.v;
        }
    }
    // ones B-matrix fragment (bf16 1.0) for WMMA row sums
    v16bf onesf;
    {
        BFV t;
#pragma unroll
        for (int j = 0; j < 8; ++j) t.u[j] = 0x3F803F80u;
        onesf = t.v;
    }

    const v8f vzero = {0.f, 0.f, 0.f, 0.f, 0.f, 0.f, 0.f, 0.f};
    v8f o[4];
    float mrow[8], lrow[8];
#pragma unroll
    for (int n = 0; n < 4; ++n) o[n] = vzero;
#pragma unroll
    for (int i = 0; i < 8; ++i) { mrow[i] = -1e30f; lrow[i] = 0.f; }

    const int srow_ = tid >> 3;  // 0..31: logical key (K) / dim (V)
    const int schk  = tid & 7;   // 0..7 : 8-element 16B chunk
    // permuted physical K rows for logical keys srow_ and srow_+32
    const int pk0 = (srow_ >> 2) + (srow_ & 3) * 16;
    const int pk1 = ((srow_ + 32) >> 2) + (srow_ & 3) * 16;

    for (int kt = 0; kt < S / CK; ++kt) {
        const size_t krow0 = (size_t)(b * S + kt * CK);
        // ---- K chunk (64 keys x 64 dims), permuted row placement
        ASYNC_B128(lds_off(&Kl[pk0 * LDK + schk * 8]),
                   qk + (krow0 + srow_) * D2 + 1024 + h * 64 + schk * 8);
        ASYNC_B128(lds_off(&Kl[pk1 * LDK + schk * 8]),
                   qk + (krow0 + srow_ + 32) * D2 + 1024 + h * 64 + schk * 8);
        // ---- V chunk (64 dims x 64 keys): contiguous copy from vT
        ASYNC_B128(lds_off(&Vl[srow_ * LDV + schk * 8]),
                   vT + (size_t)(h * 64 + srow_) * NT + krow0 + schk * 8);
        ASYNC_B128(lds_off(&Vl[(srow_ + 32) * LDV + schk * 8]),
                   vT + (size_t)(h * 64 + srow_ + 32) * NT + krow0 + schk * 8);
        WAIT_ASYNC();
        __syncthreads();

        // ---- scores over permuted keys: 4 tiles, Dh=64 -> 2 WMMAs each
        v8f s[4];
#pragma unroll
        for (int t = 0; t < 4; ++t) {
            v8f sc = vzero;
#pragma unroll
            for (int dc = 0; dc < 2; ++dc) {
                const uint16_t* p = Kl + (t * 16 + l15) * LDK + dc * 32 + kb;
                v16bf kf = frag_from(*(const uint4*)p, *(const uint4*)(p + 8));
                sc = wmma_bf16(qf[dc], kf, sc);
            }
            s[t] = sc;
        }

        // ---- online softmax: max butterfly only; sums via WMMA below
        float alpha[8];
#pragma unroll
        for (int i = 0; i < 8; ++i) {
            float m = fmaxf(fmaxf(s[0][i], s[1][i]), fmaxf(s[2][i], s[3][i]));
            m = fmaxf(m, __shfl_xor(m, 1, 32));
            m = fmaxf(m, __shfl_xor(m, 2, 32));
            m = fmaxf(m, __shfl_xor(m, 4, 32));
            m = fmaxf(m, __shfl_xor(m, 8, 32));
            float mnew = fmaxf(mrow[i], m);
            alpha[i] = __expf(mrow[i] - mnew);
            mrow[i] = mnew;
            float p0 = __expf(s[0][i] - mnew);
            float p1 = __expf(s[1][i] - mnew);
            float p2 = __expf(s[2][i] - mnew);
            float p3 = __expf(s[3][i] - mnew);
            o[0][i] *= alpha[i]; o[1][i] *= alpha[i];
            o[2][i] *= alpha[i]; o[3][i] *= alpha[i];
            // P slots for this lane are contiguous: l15*4 + t  -> one b64 store
            int r = i + hlf * 8;
            uint2 pp2;
            pp2.x = pack2bf(p0, p1);
            pp2.y = pack2bf(p2, p3);
            *(uint2*)(Pw + r * LDP + l15 * 4) = pp2;
        }

        // ---- P fragments (A layout, 16x64 -> two 16x32 fragments)
        v16bf pf[2];
#pragma unroll
        for (int kc = 0; kc < 2; ++kc) {
            const uint16_t* pp = Pw + l15 * LDP + kc * 32 + kb;
            pf[kc] = frag_from(*(const uint4*)pp, *(const uint4*)(pp + 8));
        }

        // ---- row sums via WMMA: rowsum(P) lands in each lane's own element
        v8f psum = wmma_bf16(pf[0], onesf, vzero);
        psum = wmma_bf16(pf[1], onesf, psum);
#pragma unroll
        for (int i = 0; i < 8; ++i) lrow[i] = lrow[i] * alpha[i] + psum[i];

        // ---- O += P V  (4 dim tiles x 2 key halves; slot order == key order)
#pragma unroll
        for (int n = 0; n < 4; ++n) {
#pragma unroll
            for (int kc = 0; kc < 2; ++kc) {
                const uint16_t* vp = Vl + (n * 16 + l15) * LDV + kc * 32 + kb;
                v16bf vf = frag_from(*(const uint4*)vp, *(const uint4*)(vp + 8));
                o[n] = wmma_bf16(pf[kc], vf, o[n]);
            }
        }
        __syncthreads();
    }

    // ---- normalize, store bf16 to attn workspace [B*S, D]
#pragma unroll
    for (int i = 0; i < 8; ++i) {
        float inv = 1.0f / lrow[i];
        int srow = q0 + i + hlf * 8;
#pragma unroll
        for (int n = 0; n < 4; ++n) {
            out[((size_t)(b * S + srow)) * D + h * 64 + n * 16 + l15] =
                (uint16_t)f2bf_u(o[n][i] * inv);
        }
    }
}

// ---------------------------------------------------------------------------
extern "C" void kernel_launch(void* const* d_in, const int* in_sizes, int n_in,
                              void* d_out, int out_size, void* d_ws, size_t ws_size,
                              hipStream_t stream) {
    const float* x     = (const float*)d_in[0];
    const float* w_in  = (const float*)d_in[1];
    const float* b_in  = (const float*)d_in[2];
    const float* w_out = (const float*)d_in[3];
    const float* b_out = (const float*)d_in[4];
    float* out = (float*)d_out;

    // workspace layout (bf16 elements, all 16B aligned)
    uint16_t* qk    = (uint16_t*)d_ws;                 // 8192*2048 (Q|K, token-major)
    uint16_t* vT    = qk    + (size_t)8192 * 2048;     // 1024*8192 (V, dim-major)
    uint16_t* xbf   = vT    + (size_t)1024 * 8192;     // 8192*1024
    uint16_t* wint  = xbf   + (size_t)8192 * 1024;     // 3072*1024 (w_in^T)
    uint16_t* woutt = wint  + (size_t)3072 * 1024;     // 1024*1024 (w_out^T)
    uint16_t* attnb = woutt + (size_t)1024 * 1024;     // 8192*1024

    dim3 blk(256, 1, 1);

    // 0) one-time convert / transpose-convert to bf16
    conv_bf16_kernel<<<dim3((8192 * 1024 / 4 + 255) / 256), blk, 0, stream>>>(
        x, xbf, (long)8192 * 1024 / 4);
    transpose_conv_kernel<<<dim3(3072 / 32, 1024 / 32), blk, 0, stream>>>(
        w_in, wint, 1024, 3072);
    transpose_conv_kernel<<<dim3(1024 / 32, 1024 / 32), blk, 0, stream>>>(
        w_out, woutt, 1024, 1024);

    // 1) qkv = x @ w_in + b_in  (Q/K token-major bf16; V dim-major bf16)
    gemm_bf16_kernel<1><<<dim3(3072 / 128, 8192 / 128), blk, 0, stream>>>(
        xbf, wint, b_in, (void*)qk, vT, 8192, 3072, 1024);
    // 2) flash attention (64-key chunks, all-async staging, WMMA row sums)
    attn_kernel<<<dim3(2048 / 128, 16, 4), blk, 0, stream>>>(qk, vT, attnb);
    // 3) out = attn @ w_out + b_out  (f32 out)
    gemm_bf16_kernel<0><<<dim3(1024 / 128, 8192 / 128), blk, 0, stream>>>(
        attnb, woutt, b_out, (void*)out, nullptr, 8192, 1024, 1024);
}
